// cgc_68831145886180
// MI455X (gfx1250) — matrix operand order repaired and checked by reading, not compiled, not run on hardware
//
#include <hip/hip_runtime.h>
#include <hip/hip_bf16.h>

// ---------------------------------------------------------------------------
// Problem constants (from reference)
// ---------------------------------------------------------------------------
#define NN      100000          // nodes
#define NE      1200000         // edges
#define FF      64              // node channels
#define DD      32              // edge feature dim
#define ZZ      160             // 2F + D
#define GG      64              // graphs
#define EPSV    1e-5f
#define SLOPE   0.01f

#define TILES           (NE / 16)       // 75000 edge tiles of 16
#define EDGE_THREADS    128             // 4 waves / block
#define WAVES_PER_BLOCK (EDGE_THREADS / 32)
#define EDGE_BLOCKS     512             // 2048 waves total, grid-stride

typedef __attribute__((ext_vector_type(16))) __bf16 v16bf_t;
typedef __attribute__((ext_vector_type(8)))  __bf16 v8bf_t;
typedef __attribute__((ext_vector_type(4)))  __bf16 v4bf_t;
typedef __attribute__((ext_vector_type(8)))  float  v8f_t;

// ---------------------------------------------------------------------------
// f32 -> bf16 streaming convert (4 elements / thread), count % 4 == 0
// ---------------------------------------------------------------------------
__global__ void cgc_cvt_bf16_kernel(const float* __restrict__ in,
                                    __bf16* __restrict__ out, int count4)
{
    const int i = blockIdx.x * blockDim.x + threadIdx.x;
    if (i < count4) {
        const float4 f = ((const float4*)in)[i];
        v4bf_t o;
        o[0] = (__bf16)f.x; o[1] = (__bf16)f.y;
        o[2] = (__bf16)f.z; o[3] = (__bf16)f.w;
        ((v4bf_t*)out)[i] = o;
    }
}

// ---------------------------------------------------------------------------
// Edge-message kernel: for each 16-edge tile compute
//   m = sigmoid(z @ Wf + bf) * softplus(z @ Ws + bs),  z = [x_dst, x_src, e]
// with v_wmma_f32_16x16x32_bf16 on pre-converted bf16 inputs, then
// atomically scatter-add into pre[dst]. pre must already hold the residual.
// ---------------------------------------------------------------------------
__global__ __launch_bounds__(EDGE_THREADS)
void cgc_edge_msg_kernel(const __bf16* __restrict__ xb,     // [N,F] bf16
                         const int*    __restrict__ srcIdx,
                         const int*    __restrict__ dstIdx,
                         const __bf16* __restrict__ attrb,  // [E,D] bf16
                         const float*  __restrict__ Wf,
                         const float*  __restrict__ bfv,
                         const float*  __restrict__ Ws,
                         const float*  __restrict__ bsv,
                         float*        __restrict__ pre)
{
    // Weights staged once per block as bf16 (row-major [ZZ][FF]); biases f32.
    __shared__ __bf16 lWf[ZZ * FF];
    __shared__ __bf16 lWs[ZZ * FF];
    __shared__ float  lbf[FF];
    __shared__ float  lbs[FF];

    for (int i = threadIdx.x; i < ZZ * FF; i += blockDim.x) {
        lWf[i] = (__bf16)Wf[i];
        lWs[i] = (__bf16)Ws[i];
    }
    for (int i = threadIdx.x; i < FF; i += blockDim.x) {
        lbf[i] = bfv[i];
        lbs[i] = bsv[i];
    }
    __syncthreads();

    const int lane = threadIdx.x & 31;
    const int wave = threadIdx.x >> 5;
    const int half = lane >> 4;      // 0 / 1 (lane group)
    const int mrow = lane & 15;      // A-matrix row index / D-matrix N index
    const int totalWaves = gridDim.x * WAVES_PER_BLOCK;

    for (int tile = blockIdx.x * WAVES_PER_BLOCK + wave; tile < TILES;
         tile += totalWaves) {
        const int ebase = tile * 16;

        // This lane supplies A-matrix row `mrow` (edge ebase+mrow).
        const int aedge = ebase + mrow;
        const int asrc  = srcIdx[aedge];
        const int adst  = dstIdx[aedge];

        v8f_t accF[4], accS[4];
        #pragma unroll
        for (int t = 0; t < 4; ++t)
            #pragma unroll
            for (int r = 0; r < 8; ++r) { accF[t][r] = 0.0f; accS[t][r] = 0.0f; }

        // K loop: Z = 160 = 5 chunks of 32 (bf16 WMMA depth).
        #pragma unroll
        for (int ks = 0; ks < 5; ++ks) {
            // 16-bit A 16x32 layout: element e -> k = ks*32 + (e&7) + ((e>>3)<<4) + half*8
            // => two contiguous 8-element bf16 runs (16B each) at zk0 and zk0+16.
            v8bf_t run[2];
            const int zk0 = ks * 32 + half * 8;
            #pragma unroll
            for (int rr = 0; rr < 2; ++rr) {
                const int zk = zk0 + rr * 16;
                const __bf16* p;
                if (zk < FF)            p = xb    + (size_t)adst  * FF + zk;
                else if (zk < 2 * FF)   p = xb    + (size_t)asrc  * FF + (zk - FF);
                else                    p = attrb + (size_t)aedge * DD + (zk - 2 * FF);
                run[rr] = *(const v8bf_t*)p;
            }
            const v16bf_t a = __builtin_shufflevector(
                run[0], run[1], 0, 1, 2, 3, 4, 5, 6, 7,
                8, 9, 10, 11, 12, 13, 14, 15);

            // 16-bit B 32x16 layout: lane holds column N = lane&15; element e -> K = ks*32 + (e&7) + ((e>>3)<<4) + half*8.
            const __bf16* wfrow = lWf + (size_t)(ks * 32 + half * 8) * FF + mrow;
            const __bf16* wsrow = lWs + (size_t)(ks * 32 + half * 8) * FF + mrow;
            #pragma unroll
            for (int t = 0; t < 4; ++t) {
                v16bf_t bF, bS;
                #pragma unroll
                for (int s = 0; s < 16; ++s) {
                    const int kk = (s & 7) + ((s >> 3) << 4);
                    bF[s] = wfrow[kk * FF + t * 16];
                    bS[s] = wsrow[kk * FF + t * 16];
                }
                accF[t] = __builtin_amdgcn_wmma_f32_16x16x32_bf16(
                    false, a, false, bF, (short)0, accF[t], false, false);
                accS[t] = __builtin_amdgcn_wmma_f32_16x16x32_bf16(
                    false, a, false, bS, (short)0, accS[t], false, false);
            }
        }

        // D layout: lane -> N = (lane&15) within tile t; VGPR r -> M = r + 8*half.
        #pragma unroll
        for (int r = 0; r < 8; ++r) {
            const int e = ebase + r + 8 * half;
            const int d = dstIdx[e];
            float* orow = pre + (size_t)d * FF;
            #pragma unroll
            for (int t = 0; t < 4; ++t) {
                const int n = t * 16 + mrow;
                const float vf  = accF[t][r] + lbf[n];
                const float vs  = accS[t][r] + lbs[n];
                const float sig = 1.0f / (1.0f + __expf(-vf));
                const float sp  = (vs > 20.0f) ? vs : __logf(1.0f + __expf(vs));
                unsafeAtomicAdd(orow + n, sig * sp);
            }
        }
    }
}

// ---------------------------------------------------------------------------
// GraphNorm helpers (per-graph stats via L2-resident f32 atomics)
// ---------------------------------------------------------------------------
__global__ void cgc_node_count_kernel(const int* __restrict__ batch,
                                      float* __restrict__ cnt)
{
    const int i = blockIdx.x * blockDim.x + threadIdx.x;
    if (i < NN) unsafeAtomicAdd(cnt + batch[i], 1.0f);
}

__global__ void cgc_feat_sum_kernel(const float* __restrict__ pre,
                                    const int*   __restrict__ batch,
                                    float*       __restrict__ sum)
{
    const int idx = blockIdx.x * blockDim.x + threadIdx.x;
    if (idx < NN * FF) {
        const int node = idx >> 6;
        const int c    = idx & 63;
        unsafeAtomicAdd(sum + batch[node] * FF + c, pre[idx]);
    }
}

__global__ void cgc_var_sum_kernel(const float* __restrict__ pre,
                                   const int*   __restrict__ batch,
                                   const float* __restrict__ sum,
                                   const float* __restrict__ cnt,
                                   const float* __restrict__ ms,
                                   float*       __restrict__ var)
{
    const int idx = blockIdx.x * blockDim.x + threadIdx.x;
    if (idx < NN * FF) {
        const int node = idx >> 6;
        const int c    = idx & 63;
        const int b    = batch[node];
        const float cc   = fmaxf(cnt[b], 1.0f);
        const float mean = sum[b * FF + c] / cc;
        const float t    = pre[idx] - mean * ms[c];
        unsafeAtomicAdd(var + b * FF + c, t * t);
    }
}

// Finalize: y = leaky(w * t / std + b). Writes f32 output and (optionally)
// a fused bf16 copy for the next layer's WMMA gather.
__global__ void cgc_norm_final_kernel(const float* __restrict__ pre,
                                      const int*   __restrict__ batch,
                                      const float* __restrict__ sum,
                                      const float* __restrict__ cnt,
                                      const float* __restrict__ var,
                                      const float* __restrict__ ms,
                                      const float* __restrict__ gw,
                                      const float* __restrict__ gb,
                                      float*       __restrict__ out,
                                      __bf16*      __restrict__ outb) // may be null
{
    const int idx = blockIdx.x * blockDim.x + threadIdx.x;
    if (idx < NN * FF) {
        const int node = idx >> 6;
        const int c    = idx & 63;
        const int b    = batch[node];
        const float cc   = fmaxf(cnt[b], 1.0f);
        const float mean = sum[b * FF + c] / cc;
        const float t    = pre[idx] - mean * ms[c];
        const float sd   = sqrtf(var[b * FF + c] / cc + EPSV);
        const float y    = gw[c] * t / sd + gb[c];
        const float r    = (y >= 0.0f) ? y : SLOPE * y;
        out[idx] = r;
        if (outb) outb[idx] = (__bf16)r;
    }
}

// ---------------------------------------------------------------------------
// Host-side orchestration
// ---------------------------------------------------------------------------
extern "C" void kernel_launch(void* const* d_in, const int* in_sizes, int n_in,
                              void* d_out, int out_size, void* d_ws, size_t ws_size,
                              hipStream_t stream)
{
    (void)in_sizes; (void)n_in; (void)out_size; (void)ws_size;

    const float* x     = (const float*)d_in[0];
    const int*   index = (const int*)  d_in[1];   // [2,E]: row0 = src, row1 = dst
    const float* attr  = (const float*)d_in[2];
    const int*   batch = (const int*)  d_in[3];
    const float* Wf1 = (const float*)d_in[4];
    const float* bf1 = (const float*)d_in[5];
    const float* Ws1 = (const float*)d_in[6];
    const float* bs1 = (const float*)d_in[7];
    const float* gw1 = (const float*)d_in[8];
    const float* gb1 = (const float*)d_in[9];
    const float* gm1 = (const float*)d_in[10];
    const float* Wf2 = (const float*)d_in[11];
    const float* bf2 = (const float*)d_in[12];
    const float* Ws2 = (const float*)d_in[13];
    const float* bs2 = (const float*)d_in[14];
    const float* gw2 = (const float*)d_in[15];
    const float* gb2 = (const float*)d_in[16];
    const float* gm2 = (const float*)d_in[17];

    const int* srcIdx = index;
    const int* dstIdx = index + NE;

    float* out = (float*)d_out;

    // Workspace layout:
    //   pre   : N*F f32   (CGConv pre-norm accumulator)
    //   h1    : N*F f32   (layer-1 output, f32)
    //   sum   : G*F f32, cnt : G f32, var : G*F f32 (contiguous stats region)
    //   xb    : N*F bf16  (x converted)
    //   h1b   : N*F bf16  (layer-1 output converted, fused into finalize)
    //   attrb : E*D bf16  (attr converted, shared by both layers)
    float*  ws    = (float*)d_ws;
    float*  pre   = ws;
    float*  h1    = pre + (size_t)NN * FF;
    float*  sum   = h1 + (size_t)NN * FF;
    float*  cnt   = sum + GG * FF;
    float*  var   = cnt + GG;
    __bf16* xb    = (__bf16*)(var + GG * FF);
    __bf16* h1b   = xb + (size_t)NN * FF;
    __bf16* attrb = h1b + (size_t)NN * FF;
    const size_t statsBytes = (size_t)(2 * GG * FF + GG) * sizeof(float);

    const int nfBlocks   = (NN * FF + 255) / 256;          // 25000
    const int nodeBlocks = (NN + 255) / 256;
    const int xCvtBlocks = (NN * FF / 4 + 255) / 256;      // 6250
    const int aCvtBlocks = (NE * DD / 4 + 255) / 256;      // 37500

    // ---------------- one-time converts (per call) ----------------
    cgc_cvt_bf16_kernel<<<xCvtBlocks, 256, 0, stream>>>(x, xb, NN * FF / 4);
    cgc_cvt_bf16_kernel<<<aCvtBlocks, 256, 0, stream>>>(attr, attrb, NE * DD / 4);

    // ---------------- Layer 1 ----------------
    hipMemcpyAsync(pre, x, (size_t)NN * FF * sizeof(float),
                   hipMemcpyDeviceToDevice, stream);
    hipMemsetAsync(sum, 0, statsBytes, stream);

    cgc_edge_msg_kernel<<<EDGE_BLOCKS, EDGE_THREADS, 0, stream>>>(
        xb, srcIdx, dstIdx, attrb, Wf1, bf1, Ws1, bs1, pre);

    cgc_node_count_kernel<<<nodeBlocks, 256, 0, stream>>>(batch, cnt);
    cgc_feat_sum_kernel<<<nfBlocks, 256, 0, stream>>>(pre, batch, sum);
    cgc_var_sum_kernel<<<nfBlocks, 256, 0, stream>>>(pre, batch, sum, cnt, gm1, var);
    cgc_norm_final_kernel<<<nfBlocks, 256, 0, stream>>>(
        pre, batch, sum, cnt, var, gm1, gw1, gb1, h1, h1b);

    // ---------------- Layer 2 ----------------
    hipMemcpyAsync(pre, h1, (size_t)NN * FF * sizeof(float),
                   hipMemcpyDeviceToDevice, stream);
    hipMemsetAsync(sum, 0, statsBytes, stream);

    cgc_edge_msg_kernel<<<EDGE_BLOCKS, EDGE_THREADS, 0, stream>>>(
        h1b, srcIdx, dstIdx, attrb, Wf2, bf2, Ws2, bs2, pre);

    cgc_node_count_kernel<<<nodeBlocks, 256, 0, stream>>>(batch, cnt);
    cgc_feat_sum_kernel<<<nfBlocks, 256, 0, stream>>>(pre, batch, sum);
    cgc_var_sum_kernel<<<nfBlocks, 256, 0, stream>>>(pre, batch, sum, cnt, gm2, var);
    cgc_norm_final_kernel<<<nfBlocks, 256, 0, stream>>>(
        pre, batch, sum, cnt, var, gm2, gw2, gb2, out, (__bf16*)nullptr);
}
